// LSHAttention_15539191677474
// MI455X (gfx1250) — matrix operand order repaired and checked by reading, not compiled, not run on hardware
//
#include <hip/hip_runtime.h>
#include <hip/hip_bf16.h>

typedef __attribute__((ext_vector_type(16))) __bf16 v16bf;
typedef __attribute__((ext_vector_type(8)))  float  v8f;

#define B_    2
#define S_    4096
#define HID_  1024
#define H_    8
#define D_    64
#define E_    512      // H_*D_
#define C_    64
#define NH_   4
#define NB_   128
#define L_    16384    // NH_*S_
#define NC_   256      // L_/C_
#define BH_   16       // B_*H_

// ---------------------------------------------------------------------------
// CDNA5 async global->LDS copy helpers (ASYNCcnt path)
// ---------------------------------------------------------------------------
__device__ __forceinline__ unsigned lds_addr(const void* p) {
#if defined(__HIP_DEVICE_COMPILE__)
  return (unsigned)(unsigned long long)(const __attribute__((address_space(3))) char*)p;
#else
  (void)p; return 0u;
#endif
}
__device__ __forceinline__ void async_copy_b128(unsigned lds_off, const void* gsrc) {
  unsigned long long ga = (unsigned long long)gsrc;
  asm volatile("global_load_async_to_lds_b128 %0, %1, off"
               :: "v"(lds_off), "v"(ga) : "memory");
}
__device__ __forceinline__ void wait_async0() {
  asm volatile("s_wait_asynccnt 0x0" ::: "memory");
}
__device__ __forceinline__ void wait_async3() {   // prev stage (3 ops) retired
  asm volatile("s_wait_asynccnt 0x3" ::: "memory");
}

// ---------------------------------------------------------------------------
// 1) LayerNorm (fp32 in) -> bf16 activations
// ---------------------------------------------------------------------------
__global__ void ln_kernel(const float* __restrict__ x, const float* __restrict__ w,
                          const float* __restrict__ bb, __bf16* __restrict__ y) {
  __shared__ float red[256], red2[256];
  const int row = blockIdx.x, tid = threadIdx.x;
  const float* xr = x + (size_t)row * HID_;
  float s = 0.f, s2 = 0.f;
  for (int c = tid; c < HID_; c += 256) { float v = xr[c]; s += v; s2 += v * v; }
  red[tid] = s; red2[tid] = s2; __syncthreads();
  for (int o = 128; o > 0; o >>= 1) {
    if (tid < o) { red[tid] += red[tid + o]; red2[tid] += red2[tid + o]; }
    __syncthreads();
  }
  const float mu  = red[0] * (1.f / HID_);
  const float var = red2[0] * (1.f / HID_) - mu * mu;
  const float inv = rsqrtf(var + 1e-12f);
  for (int c = tid; c < HID_; c += 256) {
    float v = (xr[c] - mu) * inv * w[c] + bb[c];
    y[(size_t)row * HID_ + c] = (__bf16)v;
  }
}

// ---------------------------------------------------------------------------
// 2) Weight swizzle: fp32 [E,HID] -> bf16 blocks [kkblk][E][32]
// ---------------------------------------------------------------------------
__global__ void wt_kernel(const float* __restrict__ wqk, const float* __restrict__ wv,
                          __bf16* __restrict__ wqkS, __bf16* __restrict__ wvS) {
  int idx = blockIdx.x * 256 + threadIdx.x;
  if (idx >= E_ * HID_) return;
  int e = idx / HID_, k = idx % HID_;
  size_t dst = ((size_t)(k >> 5) * E_ + e) * 32 + (k & 31);
  wqkS[dst] = (__bf16)wqk[idx];
  wvS[dst]  = (__bf16)wv[idx];
}

// ---------------------------------------------------------------------------
// 3) Projection GEMM (WMMA bf16), double-buffered async global->LDS pipeline,
//    LDS-transposed coalesced epilogue.
//    which==0: qk fp32 (hashing) + qk bf16; which==1: v bf16 only.
//    dynamic LDS: a[2]=10240B, b[2]=20480B (30720), epilogue C aliases = 33792B
// ---------------------------------------------------------------------------
__global__ void proj_kernel(const __bf16* __restrict__ A, const __bf16* __restrict__ wqkS,
                            const __bf16* __restrict__ wvS, float* __restrict__ qk,
                            __bf16* __restrict__ qkbf, __bf16* __restrict__ vbf) {
  extern __shared__ char psm[];
  __bf16* aL = (__bf16*)psm;             // [2][64*40]
  __bf16* bL = (__bf16*)(psm + 10240);   // [2][128*40]
  float*  cL = (float*)psm;              // [64][132] epilogue, aliases tiles

  const int tid = threadIdx.x;
  const int wave = tid >> 5, lane = tid & 31;
  const int nloc = lane & 15, half = lane >> 4;
  const int mblk = blockIdx.x, nblk = blockIdx.y, which = blockIdx.z;
  const __bf16* Bw = which ? wvS : wqkS;
  const int arow = tid >> 2, aseg = tid & 3;

  auto issue = [&](int stage, int buf) {
    async_copy_b128(lds_addr(&aL[buf * 2560 + arow * 40 + aseg * 8]),
                    A + (size_t)(mblk * 64 + arow) * HID_ + stage * 32 + aseg * 8);
#pragma unroll
    for (int u = 0; u < 2; ++u) {
      int c2 = tid * 2 + u, col = c2 >> 2, seg = c2 & 3;
      async_copy_b128(lds_addr(&bL[buf * 5120 + col * 40 + seg * 8]),
                      Bw + ((size_t)stage * E_ + nblk * 128 + col) * 32 + seg * 8);
    }
  };

  v8f acc[4] = {};
  issue(0, 0);
  for (int it = 0; it < HID_ / 32; ++it) {
    const int cur = it & 1;
    if (it + 1 < HID_ / 32) { issue(it + 1, cur ^ 1); wait_async3(); }
    else                    { wait_async0(); }
    __syncthreads();
    const __bf16* ab = &aL[cur * 2560];
    const __bf16* bb2 = &bL[cur * 5120];
    v16bf bfrag;
#pragma unroll
    for (int j = 0; j < 8; ++j) {
      int kb = half * 16 + 2 * j;                 // contiguous 16 K @ fixed col
      bfrag[2 * j]     = bb2[(wave * 16 + nloc) * 40 + kb];
      bfrag[2 * j + 1] = bb2[(wave * 16 + nloc) * 40 + kb + 1];
    }
#pragma unroll
    for (int mt = 0; mt < 4; ++mt) {
      v16bf afrag;
#pragma unroll
      for (int j = 0; j < 8; ++j) {
        int kb = half * 16 + 2 * j;
        afrag[2 * j]     = ab[(mt * 16 + nloc) * 40 + kb];
        afrag[2 * j + 1] = ab[(mt * 16 + nloc) * 40 + kb + 1];
      }
      acc[mt] = __builtin_amdgcn_wmma_f32_16x16x32_bf16(false, afrag, false, bfrag,
                                                        (short)0, acc[mt], false, false);
    }
    __syncthreads();
  }

  // LDS-transposed epilogue: C tile -> contiguous global runs
#pragma unroll
  for (int mt = 0; mt < 4; ++mt)
#pragma unroll
    for (int r = 0; r < 8; ++r)
      cL[(mt * 16 + r + 8 * half) * 132 + wave * 16 + nloc] = acc[mt][r];
  __syncthreads();

  const int row = tid >> 2, col0 = (tid & 3) * 32;   // 32 contiguous cols/thread
  const int m = mblk * 64 + row, e0 = nblk * 128 + col0;
  const int b = m >> 12, s = m & (S_ - 1), h = e0 >> 6, d0 = e0 & 63;
  const size_t gbase = ((size_t)(b * H_ + h) * S_ + s) * D_ + d0;
  const float* srcc = &cL[row * 132 + col0];
  if (which == 0) {
#pragma unroll
    for (int u = 0; u < 32; ++u) {
      float val = srcc[u];
      qk[gbase + u] = val;
      qkbf[gbase + u] = (__bf16)val;
    }
  } else {
#pragma unroll
    for (int u = 0; u < 32; ++u) vbf[gbase + u] = (__bf16)srcc[u];
  }
}

// ---------------------------------------------------------------------------
// 4) Hashing: argmax over concat(rot,-rot) -> packed u64 sort key
// ---------------------------------------------------------------------------
__global__ void hash_kernel(const float* __restrict__ qk, const float* __restrict__ amask,
                            const float* __restrict__ rotations,
                            unsigned long long* __restrict__ key64) {
  extern __shared__ float hsm[];
  float* qrow = hsm;             // [64][64]
  float* rot  = hsm + 64 * 64;   // [64][256]
  const int tid = threadIdx.x;
  const int bid = blockIdx.x;
  const int b = bid >> 9, h = (bid >> 6) & 7, sblk = bid & 63;
  const float* src = qk + ((size_t)(b * H_ + h) * S_ + sblk * 64) * D_;
  for (int i = tid; i < 64 * 64; i += 256)  qrow[i] = src[i];
  for (int i = tid; i < 64 * 256; i += 256) rot[i] = rotations[i];
  __syncthreads();
  const int t = tid & 63, nh = tid >> 6;
  float bmax = -3.4e38f, bmin = 3.4e38f;
  int imax = 0, imin = 0;
  for (int r = 0; r < 64; ++r) {
    float a = 0.f;
#pragma unroll 8
    for (int d = 0; d < 64; ++d) a += qrow[t * 64 + d] * rot[d * 256 + nh * 64 + r];
    if (a > bmax) { bmax = a; imax = r; }
    if (a < bmin) { bmin = a; imin = r; }
  }
  const int sg = sblk * 64 + t;
  int bucket = (bmax >= -bmin) ? imax : 64 + imin;
  if (amask[b * S_ + sg] <= 0.5f) bucket = NB_;
  bucket += nh * (NB_ + 1);
  const int i = nh * S_ + sg;
  unsigned long long scaled = (unsigned long long)(unsigned)(S_ * bucket + sg);
  key64[(size_t)(b * H_ + h) * L_ + i] = (scaled << 14) | (unsigned long long)i;
}

// ---------------------------------------------------------------------------
// 5) Bitonic sort of 16384 u64 keys per (b,h) in 128KB LDS
// ---------------------------------------------------------------------------
__global__ void sort_kernel(const unsigned long long* __restrict__ key64,
                            int* __restrict__ st_idx, int* __restrict__ undo) {
  extern __shared__ unsigned long long keys[];
  const int bh = blockIdx.x, tid = threadIdx.x;
  const unsigned long long* src = key64 + (size_t)bh * L_;
  for (int i = tid; i < L_; i += 1024) keys[i] = src[i];
  __syncthreads();
  for (int k = 2; k <= L_; k <<= 1) {
    for (int j = k >> 1; j > 0; j >>= 1) {
      for (int i = tid; i < L_; i += 1024) {
        int ixj = i ^ j;
        if (ixj > i) {
          bool up = (i & k) == 0;
          unsigned long long a = keys[i], c = keys[ixj];
          if (up ? (a > c) : (a < c)) { keys[i] = c; keys[ixj] = a; }
        }
      }
      __syncthreads();
    }
  }
  for (int i = tid; i < L_; i += 1024) {
    int orig = (int)(keys[i] & 0x3FFFull);
    st_idx[(size_t)bh * L_ + i] = orig & (S_ - 1);
    undo[(size_t)bh * L_ + orig] = i;
  }
}

// ---------------------------------------------------------------------------
// 6) Fused chunked attention: async row gathers -> k-norm -> QK^T (WMMA) ->
//    mask -> logsumexp -> probs -> P*V (WMMA) -> LDS-staged coalesced store.
// ---------------------------------------------------------------------------
__global__ void attn_kernel(const __bf16* __restrict__ qkbf, const __bf16* __restrict__ vbf,
                            const float* __restrict__ amask, const int* __restrict__ st_idx,
                            float* __restrict__ o_s, float* __restrict__ logits_s) {
  extern __shared__ char smem[];
  float*  dots  = (float*)smem;                  // [64][128] 32768B (reused as oc)
  __bf16* qT    = (__bf16*)(smem + 32768);       // [64][72]   9216B
  __bf16* kR    = (__bf16*)(smem + 41984);       // [128][72] 18432B
  __bf16* vT    = (__bf16*)(smem + 60416);       // [64][136] 17408B
  __bf16* probs = (__bf16*)(smem + 77824);       // [64][136] 17408B
  int*    qidx  = (int*)(smem + 95232);
  int*    kidx  = qidx + 64;
  float*  km    = (float*)(kidx + 128);          // end 96512

  const int tid = threadIdx.x;
  const int bh = blockIdx.x >> 8, n = blockIdx.x & (NC_ - 1);
  const int b = bh >> 3;
  const int base = bh * L_ + n * C_;
  const int pn = (n + NC_ - 1) & (NC_ - 1);

  if (tid < 64) qidx[tid] = st_idx[base + tid];
  if (tid >= 64 && tid < 192) {
    int k = tid - 64;
    kidx[k] = (k < 64) ? st_idx[bh * L_ + pn * C_ + k] : st_idx[base + k - 64];
  }
  __syncthreads();
  if (tid < 128) km[tid] = amask[b * S_ + kidx[tid]];

  if (tid < 64) {                                 // q rows: async 128B each
    const __bf16* srcq = qkbf + ((size_t)bh * S_ + qidx[tid]) * D_;
#pragma unroll
    for (int seg = 0; seg < 8; ++seg)
      async_copy_b128(lds_addr(&qT[tid * 72 + seg * 8]), srcq + seg * 8);
  } else if (tid < 192) {                         // k rows: async, normalize later
    int k = tid - 64;
    const __bf16* srck = qkbf + ((size_t)bh * S_ + kidx[k]) * D_;
#pragma unroll
    for (int seg = 0; seg < 8; ++seg)
      async_copy_b128(lds_addr(&kR[k * 72 + seg * 8]), srck + seg * 8);
  } else {                                        // v rows: transpose-on-write
    for (int k = tid - 192; k < 128; k += 64) {
      const __bf16* srcv = vbf + ((size_t)bh * S_ + kidx[k]) * D_;
      for (int d = 0; d < 64; ++d) vT[d * 136 + k] = srcv[d];
    }
  }
  wait_async0();
  __syncthreads();

  if (tid < 128) {                                // in-place rms-norm * D^-0.5
    float ss = 0.f;
    for (int d = 0; d < 64; ++d) { float x = (float)kR[tid * 72 + d]; ss += x * x; }
    float sc = rsqrtf(ss * (1.f / 64.f) + 1e-6f) * 0.125f;
    for (int d = 0; d < 64; ++d)
      kR[tid * 72 + d] = (__bf16)((float)kR[tid * 72 + d] * sc);
  }
  __syncthreads();

  const int wave = tid >> 5, lane = tid & 31;
  const int nloc = lane & 15, half = lane >> 4;

  // ---- dots = q[64x64] * k^T[64x128]; wave w owns key-column tile w ----
  v8f dacc[4] = {};
#pragma unroll
  for (int kk = 0; kk < 64; kk += 32) {
    v16bf bfrag;
#pragma unroll
    for (int j = 0; j < 8; ++j) {
      int kb = kk + half * 16 + 2 * j;
      bfrag[2 * j]     = kR[(wave * 16 + nloc) * 72 + kb];
      bfrag[2 * j + 1] = kR[(wave * 16 + nloc) * 72 + kb + 1];
    }
#pragma unroll
    for (int mt = 0; mt < 4; ++mt) {
      v16bf afrag;
#pragma unroll
      for (int j = 0; j < 8; ++j) {
        int ka = kk + half * 16 + 2 * j;
        afrag[2 * j]     = qT[(mt * 16 + nloc) * 72 + ka];
        afrag[2 * j + 1] = qT[(mt * 16 + nloc) * 72 + ka + 1];
      }
      dacc[mt] = __builtin_amdgcn_wmma_f32_16x16x32_bf16(false, afrag, false, bfrag,
                                                         (short)0, dacc[mt], false, false);
    }
  }
#pragma unroll
  for (int mt = 0; mt < 4; ++mt) {
#pragma unroll
    for (int r = 0; r < 8; ++r) {
      int c = mt * 16 + r + 8 * half;
      int kcol = wave * 16 + nloc;
      float val = dacc[mt][r];
      int qi = qidx[c], ki = kidx[kcol];
      if (!((qi >= ki) && (km[kcol] > 0.5f))) val = -1e9f;
      if (qi == ki) val = -1e5f;
      dots[c * 128 + kcol] = val;
    }
  }
  __syncthreads();

  if (tid < 64) {                                 // per-row logsumexp + probs
    float m = -3.4e38f;
    for (int k = 0; k < 128; ++k) m = fmaxf(m, dots[tid * 128 + k]);
    float ssum = 0.f;
    for (int k = 0; k < 128; ++k) ssum += expf(dots[tid * 128 + k] - m);
    float lg = m + logf(ssum);
    logits_s[(size_t)bh * L_ + n * C_ + tid] = lg;
    for (int k = 0; k < 128; ++k) probs[tid * 136 + k] = (__bf16)expf(dots[tid * 128 + k] - lg);
  }
  __syncthreads();

  // ---- out = probs[64x128] * v[128x64]; 16 tiles, 2 per wave ----
  float* oc = dots;                               // reuse: [64][68] f32
#pragma unroll
  for (int ti = 0; ti < 2; ++ti) {
    int t = wave + ti * 8;
    int mt = t >> 2, nt = t & 3;
    v8f oacc = {};
#pragma unroll
    for (int kk = 0; kk < 128; kk += 32) {
      v16bf bfrag, afrag;
#pragma unroll
      for (int j = 0; j < 8; ++j) {
        int kb = kk + half * 16 + 2 * j;
        bfrag[2 * j]     = vT[(nt * 16 + nloc) * 136 + kb];
        bfrag[2 * j + 1] = vT[(nt * 16 + nloc) * 136 + kb + 1];
        int ka = kk + half * 16 + 2 * j;
        afrag[2 * j]     = probs[(mt * 16 + nloc) * 136 + ka];
        afrag[2 * j + 1] = probs[(mt * 16 + nloc) * 136 + ka + 1];
      }
      oacc = __builtin_amdgcn_wmma_f32_16x16x32_bf16(false, afrag, false, bfrag,
                                                     (short)0, oacc, false, false);
    }
#pragma unroll
    for (int r = 0; r < 8; ++r)
      oc[(mt * 16 + r + 8 * half) * 68 + nt * 16 + nloc] = oacc[r];
  }
  __syncthreads();

  {                                               // coalesced o_s store
    const int row = tid >> 2, d0 = (tid & 3) * 16;
    const float* srcc = &oc[row * 68 + d0];
    float* dst = &o_s[((size_t)bh * L_ + n * C_ + row) * D_ + d0];
#pragma unroll
    for (int u = 0; u < 16; ++u) dst[u] = srcc[u];
  }
}

// ---------------------------------------------------------------------------
// 7) Undo sort + softmax-combine over hash rounds; output [B,S,H*D] fp32
// ---------------------------------------------------------------------------
__global__ void combine_kernel(const int* __restrict__ undo, const float* __restrict__ logits_s,
                               const float* __restrict__ o_s, float* __restrict__ out) {
  int gid = blockIdx.x * 256 + threadIdx.x;
  int d = gid & 63;
  int s = (gid >> 6) & (S_ - 1);
  int h = (gid >> 18) & 7;
  int b = gid >> 21;
  int bh = b * H_ + h;
  float lg[4]; int pos[4];
  float m = -3.4e38f;
#pragma unroll
  for (int nh = 0; nh < 4; ++nh) {
    pos[nh] = undo[(size_t)bh * L_ + nh * S_ + s];
    lg[nh] = logits_s[(size_t)bh * L_ + pos[nh]];
    m = fmaxf(m, lg[nh]);
  }
  float ssum = 0.f;
#pragma unroll
  for (int nh = 0; nh < 4; ++nh) { lg[nh] = expf(lg[nh] - m); ssum += lg[nh]; }
  float acc = 0.f;
#pragma unroll
  for (int nh = 0; nh < 4; ++nh)
    acc += (lg[nh] / ssum) * o_s[((size_t)bh * L_ + pos[nh]) * D_ + d];
  out[((size_t)b * S_ + s) * E_ + h * D_ + d] = acc;
}

// ---------------------------------------------------------------------------
extern "C" void kernel_launch(void* const* d_in, const int* in_sizes, int n_in,
                              void* d_out, int out_size, void* d_ws, size_t ws_size,
                              hipStream_t stream) {
  (void)in_sizes; (void)n_in; (void)out_size; (void)ws_size;
  const float* hidden    = (const float*)d_in[0];
  const float* amask     = (const float*)d_in[1];
  const float* ln_w      = (const float*)d_in[2];
  const float* ln_b      = (const float*)d_in[3];
  const float* w_qk      = (const float*)d_in[4];
  const float* w_v       = (const float*)d_in[5];
  const float* rotations = (const float*)d_in[6];
  float* out = (float*)d_out;

  char* ws = (char*)d_ws;
  size_t off = 0;
  auto take = [&](size_t bytes) -> char* {
    char* p = ws + off;
    off += bytes;
    off = (off + 255) & ~(size_t)255;
    return p;
  };
  __bf16* xln   = (__bf16*)take((size_t)B_ * S_ * HID_ * 2);
  __bf16* wqkS  = (__bf16*)take((size_t)HID_ * E_ * 2);
  __bf16* wvS   = (__bf16*)take((size_t)HID_ * E_ * 2);
  float*  qk    = (float*)take((size_t)B_ * H_ * S_ * D_ * 4);
  __bf16* qkbf  = (__bf16*)take((size_t)B_ * H_ * S_ * D_ * 2);
  __bf16* vbf   = (__bf16*)take((size_t)B_ * H_ * S_ * D_ * 2);
  unsigned long long* key64 = (unsigned long long*)take((size_t)BH_ * L_ * 8);
  int* sti   = (int*)take((size_t)BH_ * L_ * 4);
  int* undo  = (int*)take((size_t)BH_ * L_ * 4);
  float* o_s = (float*)take((size_t)BH_ * L_ * D_ * 4);
  float* lgs = (float*)take((size_t)BH_ * L_ * 4);

  ln_kernel<<<B_ * S_, 256, 0, stream>>>(hidden, ln_w, ln_b, xln);
  wt_kernel<<<(E_ * HID_ + 255) / 256, 256, 0, stream>>>(w_qk, w_v, wqkS, wvS);
  dim3 pg(128, 4, 2);
  proj_kernel<<<pg, 256, 33792, stream>>>(xln, wqkS, wvS, qk, qkbf, vbf);
  hash_kernel<<<B_ * H_ * (S_ / 64), 256, 81920, stream>>>(qk, amask, rotations, key64);
  sort_kernel<<<BH_, 1024, 131072, stream>>>(key64, sti, undo);
  attn_kernel<<<BH_ * NC_, 256, 96512, stream>>>(qkbf, vbf, amask, sti, o_s, lgs);
  combine_kernel<<<(B_ * H_ * S_ * D_) / 256, 256, 0, stream>>>(undo, lgs, o_s, out);
}
